// DeformableTransformerDecoderLayer_21543555957438
// MI455X (gfx1250) — matrix operand order repaired and checked by reading, not compiled, not run on hardware
//
#include <hip/hip_runtime.h>
#include <hip/hip_bf16.h>

// ---------------------------------------------------------------------------
// Deformable Transformer decoder layer for MI455X (gfx1250, wave32, WMMA).
// GEMMs: LDS-staged, double-buffered f16 tiles feeding v_wmma_f32_16x16x32_f16.
// Workgroup = 8 waves = 64x128 tile; each wave computes 16x64 (4 accumulators).
// K % 32 == 0 everywhere (attention key-dim padded 300 -> 320, zero-filled),
// rows/cols clamped so the hot loop is branch-free.
// ---------------------------------------------------------------------------

#define BS     8
#define NQ     300
#define NQP    320     // NQ padded to a multiple of 32 for the K dimension
#define DMODEL 256
#define NH     8
#define DH     32
#define NL     4
#define NP     4
#define DFF    1024
#define STOT   19947   // 100*150 + 50*75 + 25*38 + 13*19

#define BMT  64        // workgroup tile M
#define BNT  128       // workgroup tile N
#define BKT  32        // K step
#define LSTR 40        // LDS row stride in halves (80B: aligned + conflict-free)

typedef __attribute__((ext_vector_type(16))) _Float16 v16h;
typedef __attribute__((ext_vector_type(8)))  _Float16 v8h;
typedef __attribute__((ext_vector_type(2)))  __fp16   g2;   // cvt_pkrtz result
typedef __attribute__((ext_vector_type(8)))  float    v8f;

union H8  { v8h  v; g2 q[4]; };
union H16 { v16h v; v8h h8[2]; };
union FFrag { v8f v; float f[8]; };

__device__ __forceinline__ v8f wmma16(v16h a, v16h b, v8f c) {
  // D = A(16x32 f16) * B(32x16 f16) + C(16x16 f32)
  return __builtin_amdgcn_wmma_f32_16x16x32_f16(false, a, false, b,
                                                (short)0, c, false, false);
}

// packed f32x2 -> f16x2 (v_cvt_pk_rtz_f16_f32)
__device__ __forceinline__ void cvt4pk(g2* d, float4 s) {
  d[0] = __builtin_amdgcn_cvt_pkrtz(s.x, s.y);
  d[1] = __builtin_amdgcn_cvt_pkrtz(s.z, s.w);
}

// ---------------------------------------------------------------------------
// Generic batched GEMM: C[z] = alpha * (A[z] @ B[z]^T) + bias  (+relu)(+mask)
// A: (M,K) lda ; B: (N,K) ldb (torch-linear weight layout) ; C: (M,N) ldc.
// REQUIRES: K % 32 == 0, lda/ldb % 4 == 0, base pointers 16B aligned.
// Batch z decomposed as z1 = z/nz2, z2 = z%nz2 with independent strides.
// ---------------------------------------------------------------------------
__global__ void __launch_bounds__(256)
gemm_wmma_kernel(const float* __restrict__ A, const float* __restrict__ B,
                 const float* __restrict__ bias, void* __restrict__ Cout,
                 int M, int N, int K, int lda, int ldb, int ldc,
                 int nz2, long sA1, long sA2, long sB1, long sB2,
                 long sC1, long sC2,
                 float alpha, int relu, int store_half,
                 const unsigned char* __restrict__ mask)
{
  __shared__ __align__(16) _Float16 sA[2][BMT * LSTR];
  __shared__ __align__(16) _Float16 sB[2][BNT * LSTR];

  const int z  = blockIdx.z;
  const int z1 = z / nz2, z2 = z % nz2;
  A += (long)z1 * sA1 + (long)z2 * sA2;
  B += (long)z1 * sB1 + (long)z2 * sB2;
  const long coff = (long)z1 * sC1 + (long)z2 * sC2;

  const int tid     = threadIdx.x;
  const int lane    = tid & 31;
  const int wid     = tid >> 5;
  const int halfsel = lane >> 4;      // 0: lanes 0-15, 1: lanes 16-31
  const int l16     = lane & 15;

  const int tileM0 = blockIdx.x * BMT;
  const int tileN0 = blockIdx.y * BNT;
  const int mrow   = (wid >> 1) * 16;   // wave M strip within tile
  const int ncol   = (wid & 1)  * 64;   // wave N strip within tile

  // Staging coordinates: A 64x32 (8 f32/thread), B 128x32 (16 f32/thread).
  // Out-of-range rows/cols clamp to the last valid row; results land only in
  // outputs the epilogue discards (WMMA rows/cols are independent).
  const int ar = tid >> 2, ac = (tid & 3) * 8;
  const int bn = tid >> 1, bc = (tid & 1) * 16;
  const float* Ag = A + (long)min(tileM0 + ar, M - 1) * lda;
  const float* Bg = B + (long)min(tileN0 + bn, N - 1) * ldb;

  FFrag acc[4];
#pragma unroll
  for (int t = 0; t < 4; ++t)
#pragma unroll
    for (int i = 0; i < 8; ++i) acc[t].f[i] = 0.0f;

  auto stage = [&](int buf, int k0) {
    const float4* pa = reinterpret_cast<const float4*>(Ag + k0 + ac);
    float4 a0 = pa[0], a1 = pa[1];
    const float4* pb = reinterpret_cast<const float4*>(Bg + k0 + bc);
    float4 b0 = pb[0], b1 = pb[1], b2 = pb[2], b3 = pb[3];
    H8 ha;  cvt4pk(ha.q + 0, a0);  cvt4pk(ha.q + 2, a1);
    H8 hb0; cvt4pk(hb0.q + 0, b0); cvt4pk(hb0.q + 2, b1);
    H8 hb1; cvt4pk(hb1.q + 0, b2); cvt4pk(hb1.q + 2, b3);
    *reinterpret_cast<v8h*>(&sA[buf][ar * LSTR + ac])     = ha.v;
    *reinterpret_cast<v8h*>(&sB[buf][bn * LSTR + bc])     = hb0.v;
    *reinterpret_cast<v8h*>(&sB[buf][bn * LSTR + bc + 8]) = hb1.v;
  };

  auto compute = [&](int buf) {
    // A fragment: elem j -> k = 8*halfsel + j + 8*(j>=8)
    H16 af;
    const _Float16* pa = &sA[buf][(mrow + l16) * LSTR + 8 * halfsel];
    af.h8[0] = *reinterpret_cast<const v8h*>(pa);
    af.h8[1] = *reinterpret_cast<const v8h*>(pa + 16);
    // B fragments: elem j -> k = 8*halfsel + j + 8*(j>=8)
#pragma unroll
    for (int t = 0; t < 4; ++t) {
      const _Float16* pb = &sB[buf][(ncol + t * 16 + l16) * LSTR + 8 * halfsel];
      H16 bf;
      bf.h8[0] = *reinterpret_cast<const v8h*>(pb);
      bf.h8[1] = *reinterpret_cast<const v8h*>(pb + 16);
      acc[t].v = wmma16(af.v, bf.v, acc[t].v);
    }
  };

  // Double-buffered pipeline: next tile's global loads are issued before the
  // current tile's WMMAs, so HBM latency overlaps LDS reads + matrix math.
  stage(0, 0);
  int buf = 0;
  for (int k0 = 0; k0 < K; k0 += BKT) {
    __syncthreads();                       // staging of `buf` complete
    if (k0 + BKT < K) stage(buf ^ 1, k0 + BKT);
    compute(buf);
    buf ^= 1;
  }

  // Epilogue: acc element r -> M = tileM + r + 8*halfsel, N = tileN+16t+l16.
  const int tileM = tileM0 + mrow;
  const int tileN = tileN0 + ncol;
  const int rowoff = halfsel * 8;
#pragma unroll
  for (int r = 0; r < 8; ++r) {
    const int m = tileM + r + rowoff;
    if (m >= M) continue;
    const bool mz = (mask != nullptr) && (mask[m] != 0);
#pragma unroll
    for (int t = 0; t < 4; ++t) {
      const int n = tileN + t * 16 + l16;
      if (n >= N) continue;
      float v = acc[t].f[r] * alpha;
      if (bias) v += bias[n];
      if (relu) v = v > 0.0f ? v : 0.0f;
      if (mz)   v = 0.0f;
      const long ci = coff + (long)m * ldc + n;
      if (store_half) ((_Float16*)Cout)[ci] = (_Float16)v;
      else            ((float*)Cout)[ci]    = v;
    }
  }
}

// ---------------------------------------------------------------------------
// Elementwise add
// ---------------------------------------------------------------------------
__global__ void add_kernel(const float* __restrict__ a,
                           const float* __restrict__ b,
                           float* __restrict__ o, int n) {
  int i = blockIdx.x * blockDim.x + threadIdx.x;
  if (i < n) o[i] = a[i] + b[i];
}

// ---------------------------------------------------------------------------
// Row softmax over cols valid entries of rows with leading-dim ld; pad
// columns [cols, ld) are zeroed so they contribute nothing as a padded K-dim.
// One wave32 per row.
// ---------------------------------------------------------------------------
__global__ void __launch_bounds__(256)
softmax_rows_kernel(float* __restrict__ x, int rows, int cols, int ld) {
  const int row = blockIdx.x * 8 + (threadIdx.x >> 5);
  if (row >= rows) return;
  const int lane = threadIdx.x & 31;
  float* p = x + (long)row * ld;
  float mx = -1e30f;
  for (int i = lane; i < cols; i += 32) mx = fmaxf(mx, p[i]);
#pragma unroll
  for (int o = 16; o > 0; o >>= 1) mx = fmaxf(mx, __shfl_xor(mx, o, 32));
  float s = 0.0f;
  for (int i = lane; i < cols; i += 32) { float e = __expf(p[i] - mx); p[i] = e; s += e; }
#pragma unroll
  for (int o = 16; o > 0; o >>= 1) s += __shfl_xor(s, o, 32);
  const float inv = 1.0f / s;
  for (int i = lane; i < cols; i += 32) p[i] *= inv;
  for (int i = cols + lane; i < ld; i += 32) p[i] = 0.0f;   // zero K padding
}

// ---------------------------------------------------------------------------
// Softmax over contiguous groups of 16 (deformable attention weights).
// ---------------------------------------------------------------------------
__global__ void softmax16_kernel(float* __restrict__ x, int ngroups) {
  const int g = blockIdx.x * blockDim.x + threadIdx.x;
  if (g >= ngroups) return;
  float* p = x + (long)g * 16;
  float mx = -1e30f;
#pragma unroll
  for (int i = 0; i < 16; ++i) mx = fmaxf(mx, p[i]);
  float e[16]; float s = 0.0f;
#pragma unroll
  for (int i = 0; i < 16; ++i) { e[i] = __expf(p[i] - mx); s += e[i]; }
  const float inv = 1.0f / s;
#pragma unroll
  for (int i = 0; i < 16; ++i) p[i] = e[i] * inv;
}

// ---------------------------------------------------------------------------
// vt[(b*8+h)*DH*NQP + d*NQP + key] = v[(b*NQ+key)*D + h*DH + d], key>=NQ -> 0
// ---------------------------------------------------------------------------
__global__ void transpose_v_kernel(const float* __restrict__ v,
                                   float* __restrict__ vt, int n) {
  int i = blockIdx.x * blockDim.x + threadIdx.x;
  if (i >= n) return;
  const int z = i / (DH * NQP);
  const int r = i % (DH * NQP);
  const int d = r / NQP, key = r % NQP;
  const int b = z / NH, h = z % NH;
  vt[i] = (key < NQ) ? v[((long)(b * NQ + key)) * DMODEL + h * DH + d] : 0.0f;
}

// ---------------------------------------------------------------------------
// out = LayerNorm(a + b) * g + beta    (cols = 256, one wave32 per row)
// ---------------------------------------------------------------------------
__global__ void __launch_bounds__(256)
add_ln_kernel(const float* __restrict__ a, const float* __restrict__ b,
              const float* __restrict__ g, const float* __restrict__ beta,
              float* __restrict__ out, int rows) {
  const int row = blockIdx.x * 8 + (threadIdx.x >> 5);
  if (row >= rows) return;
  const int lane = threadIdx.x & 31;
  const float* pa = a + (long)row * DMODEL;
  const float* pb = b + (long)row * DMODEL;
  float v[8]; float s = 0.0f;
#pragma unroll
  for (int i = 0; i < 8; ++i) { int c = lane + i * 32; v[i] = pa[c] + pb[c]; s += v[i]; }
#pragma unroll
  for (int o = 16; o > 0; o >>= 1) s += __shfl_xor(s, o, 32);
  const float mean = s * (1.0f / DMODEL);
  float q = 0.0f;
#pragma unroll
  for (int i = 0; i < 8; ++i) { float d = v[i] - mean; q += d * d; }
#pragma unroll
  for (int o = 16; o > 0; o >>= 1) q += __shfl_xor(q, o, 32);
  const float rstd = rsqrtf(q * (1.0f / DMODEL) + 1e-5f);
  float* po = out + (long)row * DMODEL;
#pragma unroll
  for (int i = 0; i < 8; ++i) {
    int c = lane + i * 32;
    po[c] = (v[i] - mean) * rstd * g[c] + beta[c];
  }
}

// ---------------------------------------------------------------------------
// Multi-scale deformable sampling. Grid (NQ, BS); block 256 = 8 waves;
// wave = head, lane = channel d. value is f16 (b*S+s, 256) row-major, so the
// 32-lane gather of one (point, head) is one coalesced 64B cacheline.
// ---------------------------------------------------------------------------
__global__ void __launch_bounds__(256)
deform_kernel(const float* __restrict__ offb, const float* __restrict__ awb,
              const float* __restrict__ ref, const _Float16* __restrict__ value,
              float* __restrict__ out) {
  const int q = blockIdx.x, b = blockIdx.y;
  const int h = threadIdx.x >> 5, d = threadIdx.x & 31;
  const int HLs[NL] = {100, 50, 25, 13};
  const int WLs[NL] = {150, 75, 38, 19};
  const int S0s[NL] = {0, 15000, 18750, 19700};

  const long row = (long)b * NQ + q;
  const float* offp = offb + row * DMODEL + h * (NL * NP * 2);
  const float* awp  = awb  + row * (NH * NL * NP) + h * (NL * NP);
  const float* refp = ref  + row * NL * 2;
  const _Float16* vb = value + ((long)b * STOT) * DMODEL + h * DH + d;

  float acc = 0.0f;
#pragma unroll
  for (int l = 0; l < NL; ++l) {
    const int   Hl = HLs[l], Wl = WLs[l];
    const float Hf = (float)Hl, Wf = (float)Wl;
    const float rx = refp[l * 2 + 0], ry = refp[l * 2 + 1];
    const _Float16* vl = vb + (long)S0s[l] * DMODEL;
#pragma unroll
    for (int p = 0; p < NP; ++p) {
      const float ox = offp[(l * NP + p) * 2 + 0];
      const float oy = offp[(l * NP + p) * 2 + 1];
      const float x = (rx + ox / Wf) * Wf - 0.5f;
      const float y = (ry + oy / Hf) * Hf - 0.5f;
      const float x0f = floorf(x), y0f = floorf(y);
      const float tx = x - x0f, ty = y - y0f;
      const int x0 = (int)x0f, y0 = (int)y0f;
      const float aw = awp[l * NP + p];
      const float cw[4] = {(1.f - ty) * (1.f - tx), (1.f - ty) * tx,
                           ty * (1.f - tx),         ty * tx};
      const int dxs[4] = {0, 1, 0, 1}, dys[4] = {0, 0, 1, 1};
#pragma unroll
      for (int c = 0; c < 4; ++c) {
        const int xi = x0 + dxs[c], yi = y0 + dys[c];
        const bool valid = (xi >= 0) && (xi < Wl) && (yi >= 0) && (yi < Hl);
        const int xc = min(max(xi, 0), Wl - 1);
        const int yc = min(max(yi, 0), Hl - 1);
        const float val = (float)vl[(long)(yc * Wl + xc) * DMODEL];
        acc += aw * cw[c] * (valid ? val : 0.0f);
      }
    }
  }
  out[row * DMODEL + h * DH + d] = acc;
}

// ---------------------------------------------------------------------------
// Host side
// ---------------------------------------------------------------------------
static void gemm(hipStream_t s, const float* A, const float* B, const float* bias,
                 void* C, int M, int N, int K, int lda, int ldb, int ldc,
                 int Z, int nz2, long sA1, long sA2, long sB1, long sB2,
                 long sC1, long sC2, float alpha, int relu, int store_half,
                 const unsigned char* mask) {
  dim3 grid((M + BMT - 1) / BMT, (N + BNT - 1) / BNT, Z), block(256);
  gemm_wmma_kernel<<<grid, block, 0, s>>>(A, B, bias, C, M, N, K, lda, ldb, ldc,
                                          nz2, sA1, sA2, sB1, sB2, sC1, sC2,
                                          alpha, relu, store_half, mask);
}

extern "C" void kernel_launch(void* const* d_in, const int* in_sizes, int n_in,
                              void* d_out, int out_size, void* d_ws, size_t ws_size,
                              hipStream_t stream) {
  const float* tgt      = (const float*)d_in[0];
  const float* pos      = (const float*)d_in[1];
  const float* refpts   = (const float*)d_in[2];
  const float* memory   = (const float*)d_in[3];
  const unsigned char* pad_mask = (const unsigned char*)d_in[4];
  // d_in[5]/d_in[6]: spatial shapes / level starts -- compile-time constants.
  const float* in_w     = (const float*)d_in[7];
  const float* in_b     = (const float*)d_in[8];
  const float* out_w    = (const float*)d_in[9];
  const float* out_b    = (const float*)d_in[10];
  const float* n1g = (const float*)d_in[11], *n1b = (const float*)d_in[12];
  const float* n2g = (const float*)d_in[13], *n2b = (const float*)d_in[14];
  const float* n3g = (const float*)d_in[15], *n3b = (const float*)d_in[16];
  const float* so_w = (const float*)d_in[17], *so_b = (const float*)d_in[18];
  const float* aw_w = (const float*)d_in[19], *aw_b = (const float*)d_in[20];
  const float* v_w  = (const float*)d_in[21], *v_b  = (const float*)d_in[22];
  const float* op_w = (const float*)d_in[23], *op_b = (const float*)d_in[24];
  const float* l1_w = (const float*)d_in[25], *l1_b = (const float*)d_in[26];
  const float* l2_w = (const float*)d_in[27], *l2_b = (const float*)d_in[28];
  float* outp = (float*)d_out;

  const int ROWS = BS * NQ;            // 2400
  const int VM   = BS * STOT;          // 159576

  // ---- workspace layout (floats) ----
  float* w = (float*)d_ws; size_t o = 0;
  float* qkin  = w + o; o += (size_t)ROWS * DMODEL;        // tgt + pos
  float* qk    = w + o; o += (size_t)ROWS * 512;           // Q|K projections
  float* vbuf  = w + o; o += (size_t)ROWS * DMODEL;        // self-attn V
  float* probs = w + o; o += (size_t)BS * NH * NQ * NQP;   // attn probs (padded)
  float* vt    = w + o; o += (size_t)BS * NH * DH * NQP;   // V^T (padded keys)
  float* attnO = w + o; o += (size_t)ROWS * DMODEL;
  float* t2    = w + o; o += (size_t)ROWS * DMODEL;
  float* tgt1  = w + o; o += (size_t)ROWS * DMODEL;
  float* q2    = w + o; o += (size_t)ROWS * DMODEL;
  float* offb  = w + o; o += (size_t)ROWS * DMODEL;        // sampling offsets
  float* awb   = w + o; o += (size_t)ROWS * 128;           // attention weights
  float* msout = w + o; o += (size_t)ROWS * DMODEL;
  float* tgt2  = w + o; o += (size_t)ROWS * DMODEL;
  float* ffh   = w + o; o += (size_t)ROWS * DFF;
  _Float16* value = (_Float16*)(w + o);                    // f16: VM x 256

  const float invsq = 0.17677669529663687f;  // 1/sqrt(32)
  const int n256 = ROWS * DMODEL;

  // ---- self attention ----
  add_kernel<<<(n256 + 255) / 256, 256, 0, stream>>>(tgt, pos, qkin, n256);
  // Q|K = qkin @ in_w[0:512]^T + in_b[0:512]
  gemm(stream, qkin, in_w, in_b, qk, ROWS, 512, 256, 256, 256, 512,
       1, 1, 0, 0, 0, 0, 0, 0, 1.0f, 0, 0, nullptr);
  // V = tgt @ in_w[512:768]^T + in_b[512:]
  gemm(stream, tgt, in_w + 512 * 256, in_b + 512, vbuf, ROWS, 256, 256,
       256, 256, 256, 1, 1, 0, 0, 0, 0, 0, 0, 1.0f, 0, 0, nullptr);
  // logits[b,h] = Qh @ Kh^T / sqrt(DH)   (z = b*8 + h), ldc = NQP (padded)
  gemm(stream, qk, qk + 256, nullptr, probs, NQ, NQ, DH, 512, 512, NQP,
       BS * NH, NH, (long)NQ * 512, 32, (long)NQ * 512, 32,
       (long)NH * NQ * NQP, (long)NQ * NQP, invsq, 0, 0, nullptr);
  softmax_rows_kernel<<<(BS * NH * NQ + 7) / 8, 256, 0, stream>>>(
      probs, BS * NH * NQ, NQ, NQP);
  {
    const int nvt = BS * NH * DH * NQP;
    transpose_v_kernel<<<(nvt + 255) / 256, 256, 0, stream>>>(vbuf, vt, nvt);
  }
  // O[b,h] = P @ Vt^T over padded K=NQP  -> attnO (b,q, h*32+d)
  gemm(stream, probs, vt, nullptr, attnO, NQ, DH, NQP, NQP, NQP, 256,
       BS * NH, NH, (long)NH * NQ * NQP, (long)NQ * NQP,
       (long)NH * DH * NQP, (long)DH * NQP, (long)NQ * DMODEL, 32,
       1.0f, 0, 0, nullptr);
  gemm(stream, attnO, out_w, out_b, t2, ROWS, 256, 256, 256, 256, 256,
       1, 1, 0, 0, 0, 0, 0, 0, 1.0f, 0, 0, nullptr);
  add_ln_kernel<<<(ROWS + 7) / 8, 256, 0, stream>>>(t2, tgt, n2g, n2b, tgt1, ROWS);

  // ---- multi-scale deformable attention ----
  add_kernel<<<(n256 + 255) / 256, 256, 0, stream>>>(tgt1, pos, q2, n256);
  gemm(stream, q2, so_w, so_b, offb, ROWS, 256, 256, 256, 256, 256,
       1, 1, 0, 0, 0, 0, 0, 0, 1.0f, 0, 0, nullptr);
  gemm(stream, q2, aw_w, aw_b, awb, ROWS, 128, 256, 256, 256, 128,
       1, 1, 0, 0, 0, 0, 0, 0, 1.0f, 0, 0, nullptr);
  softmax16_kernel<<<(ROWS * NH + 255) / 256, 256, 0, stream>>>(awb, ROWS * NH);
  // value = memory @ v_w^T + v_b  (f16 store, masked rows zeroed)
  gemm(stream, memory, v_w, v_b, value, VM, 256, 256, 256, 256, 256,
       1, 1, 0, 0, 0, 0, 0, 0, 1.0f, 0, 1, pad_mask);
  {
    dim3 g(NQ, BS), blk(256);
    deform_kernel<<<g, blk, 0, stream>>>(offb, awb, refpts, value, msout);
  }
  gemm(stream, msout, op_w, op_b, t2, ROWS, 256, 256, 256, 256, 256,
       1, 1, 0, 0, 0, 0, 0, 0, 1.0f, 0, 0, nullptr);
  add_ln_kernel<<<(ROWS + 7) / 8, 256, 0, stream>>>(t2, tgt1, n1g, n1b, tgt2, ROWS);

  // ---- FFN ----
  gemm(stream, tgt2, l1_w, l1_b, ffh, ROWS, DFF, 256, 256, 256, DFF,
       1, 1, 0, 0, 0, 0, 0, 0, 1.0f, 1 /*relu*/, 0, nullptr);
  gemm(stream, ffh, l2_w, l2_b, t2, ROWS, 256, DFF, DFF, DFF, 256,
       1, 1, 0, 0, 0, 0, 0, 0, 1.0f, 0, 0, nullptr);
  add_ln_kernel<<<(ROWS + 7) / 8, 256, 0, stream>>>(t2, tgt2, n3g, n3b, outp, ROWS);
}
